// VectorQuantizer_20160576487973
// MI455X (gfx1250) — matrix operand order repaired and checked by reading, not compiled, not run on hardware
//
#include <hip/hip_runtime.h>

// ---------------- problem constants ----------------
#define DIM       256
#define KCODES    8192
#define NTOK      32768            // B*T = 16*2048
#define ROWS_BLK  128              // rows per block (8 waves x 16)
#define KT        32               // codebook cols per tile
#define LROW      264              // LDS row stride in bf16 elems (256 + 8 pad -> conflict-free)
#define DCHUNKS   8                // 256 / 32
#define NTILES    (KCODES / KT)    // 256

typedef __attribute__((ext_vector_type(16))) __bf16        v16bf;
typedef __attribute__((ext_vector_type(8)))  float         v8f;
typedef __attribute__((ext_vector_type(4)))  unsigned int  v4u;
typedef __attribute__((ext_vector_type(4)))  int           v4i;
typedef __attribute__((ext_vector_type(8)))  int           v8i;

union BF16x16 {
    v16bf v;
    unsigned short u[16];
    uint4 q[2];
};

// split float into bf16 hi (truncation) + bf16 residual lo: x ~= hi + lo, ~2^-16 rel error
static __device__ __forceinline__ void splitf(float x, unsigned short& h, unsigned short& l) {
    union { float f; unsigned u; } a; a.f = x;
    unsigned short hb = (unsigned short)(a.u >> 16);
    union { unsigned u; float f; } b; b.u = (unsigned)hb << 16;
    union { float f; unsigned u; } c; c.f = x - b.f;
    h = hb;
    l = (unsigned short)(c.u >> 16);
}

// ---------------- kernel: exact fp32 |c_k|^2 ----------------
__global__ void vq_cnorm_kernel(const float* __restrict__ cb, float* __restrict__ cnorm) {
    int k = blockIdx.x * blockDim.x + threadIdx.x;
    if (k >= KCODES) return;
    const float4* p = (const float4*)(cb + (size_t)k * DIM);
    float s = 0.f;
    #pragma unroll 4
    for (int i = 0; i < DIM / 4; ++i) {
        float4 v = p[i];
        s += v.x * v.x + v.y * v.y + v.z * v.z + v.w * v.w;
    }
    cnorm[k] = s;
}

// ---------------- kernel: one-time codebook fp32 -> bf16 hi/lo planes ----------------
__global__ void vq_prep_kernel(const float* __restrict__ cb,
                               unsigned short* __restrict__ cbhi,
                               unsigned short* __restrict__ cblo) {
    int i = blockIdx.x * blockDim.x + threadIdx.x;        // float4 index
    if (i >= KCODES * DIM / 4) return;
    float4 v = ((const float4*)cb)[i];
    unsigned short h0, h1, h2, h3, l0, l1, l2, l3;
    splitf(v.x, h0, l0); splitf(v.y, h1, l1);
    splitf(v.z, h2, l2); splitf(v.w, h3, l3);
    *(uint2*)&cbhi[i * 4] = make_uint2((unsigned)h0 | ((unsigned)h1 << 16),
                                       (unsigned)h2 | ((unsigned)h3 << 16));
    *(uint2*)&cblo[i * 4] = make_uint2((unsigned)l0 | ((unsigned)l1 << 16),
                                       (unsigned)l2 | ((unsigned)l3 << 16));
}

// ---------------- TDM: DMA one 32x256 bf16 tile (512B rows) into LDS with 16B row pad ----
static __device__ __forceinline__ void tdm_load_tile(unsigned lds_off, const unsigned short* gsrc) {
    unsigned long long ga = (unsigned long long)(uintptr_t)gsrc;
    v4u g0;
    g0.x = 1u;                                               // count=1, user descriptor
    g0.y = lds_off;                                          // lds_addr [63:32]
    g0.z = (unsigned)(ga & 0xFFFFFFFFull);                   // global_addr lo
    g0.w = (unsigned)((ga >> 32) & 0x01FFFFFFull) | (2u << 30); // global_addr hi | type=2
    v8i g1;
    // pad_enable | pad_interval=6 (every 64 dwords = 512B) | pad_amount=3 (4 dwords = 16B)
    g1[0] = (int)((1u << 20) | (6u << 22) | (3u << 25));     // data_size=0 (1B), wg_mask=0
    g1[1] = (int)(512u << 16);                               // tensor_dim0[15:0]=512 bytes
    g1[2] = (int)(8192u << 16);                              // tensor_dim0 hi=0 | tensor_dim1 lo=8192
    g1[3] = (int)(512u << 16);                               // tensor_dim1 hi=0 | tile_dim0=512
    g1[4] = KT;                                              // tile_dim1=32, tile_dim2=0
    g1[5] = 512;                                             // tensor_dim0_stride lo = 512 bytes
    g1[6] = 0;
    g1[7] = 0;
    v4i z4 = {0, 0, 0, 0};                                   // 2D tensor: groups 2/3 unused
#if defined(__clang_major__) && (__clang_major__ >= 23)
    v8i z8 = {0, 0, 0, 0, 0, 0, 0, 0};
    __builtin_amdgcn_tensor_load_to_lds(g0, g1, z4, z4, z8, 0);
#else
    __builtin_amdgcn_tensor_load_to_lds(g0, g1, z4, z4, 0);
#endif
}

// ---- shared compute helpers for both main kernels ----
struct ArgminState {
    float best[8];
    int   bidx[8];
};

static __device__ __forceinline__ void build_A_frags(const float* __restrict__ z, int row0,
                                                     int w, int lane,
                                                     BF16x16* Ahi, BF16x16* Alo) {
    // ISA A layout (16-bit 16x32): lanes 0-15: M=lane, K in {0..7,16..23};
    //                              lanes 16-31: M=lane-16, K in {8..15,24..31}
    const int m  = lane & 15;
    const int kb = (lane < 16) ? 0 : 8;
    const float* zrow = z + (size_t)(row0 + w * 16 + m) * DIM;
    #pragma unroll
    for (int d = 0; d < DCHUNKS; ++d) {
        const float* p = zrow + d * 32;
        #pragma unroll
        for (int j = 0; j < 8; ++j) splitf(p[kb + j],      Ahi[d].u[j],     Alo[d].u[j]);
        #pragma unroll
        for (int j = 0; j < 8; ++j) splitf(p[16 + kb + j], Ahi[d].u[8 + j], Alo[d].u[8 + j]);
    }
}

static __device__ __forceinline__ void compute_tile(const unsigned short* __restrict__ sBhi,
                                                    const unsigned short* __restrict__ sBlo,
                                                    const float* __restrict__ sCn,
                                                    const BF16x16* Ahi, const BF16x16* Alo,
                                                    int lane, int k0, ArgminState& st) {
    const int n    = lane & 15;
    const int koff = (lane < 16) ? 0 : 8;
    #pragma unroll
    for (int sub = 0; sub < 2; ++sub) {
        v8f acc = {0.f, 0.f, 0.f, 0.f, 0.f, 0.f, 0.f, 0.f};
        const int rowb = (sub * 16 + n) * LROW;
        #pragma unroll
        for (int d = 0; d < DCHUNKS; ++d) {
            BF16x16 bh, bl;
            const uint4* ph = (const uint4*)&sBhi[rowb + d * 32 + koff];
            bh.q[0] = ph[0]; bh.q[1] = ph[2];
            const uint4* pl = (const uint4*)&sBlo[rowb + d * 32 + koff];
            bl.q[0] = pl[0]; bl.q[1] = pl[2];
            // error-compensated fp32-ish dot: zh*ch + zh*cl + zl*ch
            acc = __builtin_amdgcn_wmma_f32_16x16x32_bf16(false, Ahi[d].v, false, bh.v,
                                                          (short)0, acc, false, false);
            acc = __builtin_amdgcn_wmma_f32_16x16x32_bf16(false, Ahi[d].v, false, bl.v,
                                                          (short)0, acc, false, false);
            acc = __builtin_amdgcn_wmma_f32_16x16x32_bf16(false, Alo[d].v, false, bh.v,
                                                          (short)0, acc, false, false);
        }
        const float cn  = sCn[sub * 16 + n];
        const int   col = k0 + sub * 16 + n;
        #pragma unroll
        for (int r = 0; r < 8; ++r) {
            float s = cn - 2.0f * acc[r];            // |c|^2 - 2 z.c (|z|^2 const per row)
            if (s < st.best[r]) { st.best[r] = s; st.bidx[r] = col; }
        }
    }
}

static __device__ __forceinline__ void finish_block(const float* __restrict__ z,
                                                    const float* __restrict__ cb,
                                                    float* __restrict__ zq_out,
                                                    int* __restrict__ idx_out,
                                                    float* __restrict__ partials,
                                                    int* sIdx, float* sRed,
                                                    int tid, int lane, int w, int row0,
                                                    ArgminState& st) {
    // cross-lane argmin within 16-lane groups (tie-break: smaller index = first hit)
    #pragma unroll
    for (int r = 0; r < 8; ++r) {
        float s = st.best[r];
        int   i = st.bidx[r];
        #pragma unroll
        for (int off = 8; off >= 1; off >>= 1) {
            float s2 = __shfl_xor(s, off, 32);
            int   i2 = __shfl_xor(i, off, 32);
            if (s2 < s || (s2 == s && i2 < i)) { s = s2; i = i2; }
        }
        if ((lane & 15) == 0) {
            int lrow = w * 16 + r + ((lane >= 16) ? 8 : 0);  // C/D layout: VGPR r -> M=r (+8 upper)
            sIdx[lrow] = i;
            idx_out[row0 + lrow] = i;
        }
    }
    __syncthreads();

    // gather z_q (exact fp32 codebook) + squared-diff partial
    float sq = 0.f;
    #pragma unroll 4
    for (int it = 0; it < (ROWS_BLK * DIM / 4) / 256; ++it) {
        int flat4 = it * 256 + tid;
        int lrow  = flat4 >> 6;
        int dim4  = flat4 & 63;
        int idx   = sIdx[lrow];
        float4 c  = ((const float4*)(cb + (size_t)idx * DIM))[dim4];
        float4 zv = ((const float4*)(z + (size_t)(row0 + lrow) * DIM))[dim4];
        ((float4*)(zq_out + (size_t)(row0 + lrow) * DIM))[dim4] = c;
        float dx = c.x - zv.x, dy = c.y - zv.y, dz2 = c.z - zv.z, dw = c.w - zv.w;
        sq += dx * dx + dy * dy + dz2 * dz2 + dw * dw;
    }
    sRed[tid] = sq;
    __syncthreads();
    #pragma unroll
    for (int s = 128; s > 0; s >>= 1) {
        if (tid < s) sRed[tid] += sRed[tid + s];
        __syncthreads();
    }
    if (tid == 0) partials[blockIdx.x] = sRed[0];
}

// ---------------- main kernel, TDM path: async DMA staging, double-buffered -------------
__global__ void __launch_bounds__(256)
vq_main_tdm_kernel(const float* __restrict__ z,
                   const float* __restrict__ cb,
                   const unsigned short* __restrict__ cbhi,
                   const unsigned short* __restrict__ cblo,
                   const float* __restrict__ cnorm,
                   float* __restrict__ zq_out,
                   int*   __restrict__ idx_out,
                   float* __restrict__ partials) {
    __shared__ unsigned short sBhi[2][KT * LROW];   // 2 x 16,896 B
    __shared__ unsigned short sBlo[2][KT * LROW];   // 2 x 16,896 B
    __shared__ float sCn[KT];
    __shared__ int   sIdx[ROWS_BLK];
    __shared__ float sRed[256];

    const int tid  = threadIdx.x;
    const int lane = tid & 31;
    const int w    = tid >> 5;
    const int row0 = blockIdx.x * ROWS_BLK;

    BF16x16 Ahi[DCHUNKS], Alo[DCHUNKS];
    build_A_frags(z, row0, w, lane, Ahi, Alo);

    ArgminState st;
    #pragma unroll
    for (int r = 0; r < 8; ++r) { st.best[r] = 3.402823466e+38f; st.bidx[r] = 0; }

    const unsigned hiOff0 = (unsigned)((uintptr_t)&sBhi[0][0]) & 0xFFFFFFFFu;
    const unsigned hiOff1 = (unsigned)((uintptr_t)&sBhi[1][0]) & 0xFFFFFFFFu;
    const unsigned loOff0 = (unsigned)((uintptr_t)&sBlo[0][0]) & 0xFFFFFFFFu;
    const unsigned loOff1 = (unsigned)((uintptr_t)&sBlo[1][0]) & 0xFFFFFFFFu;

    if (w == 0) {                                   // wave 0 drives the Tensor Data Mover
        tdm_load_tile(hiOff0, cbhi);
        tdm_load_tile(loOff0, cblo);
    }

    for (int t = 0; t < NTILES; ++t) {
        const int cur = t & 1;
        if (w == 0) {
            if (t + 1 < NTILES) {
                const size_t goff = (size_t)(t + 1) * KT * DIM;   // bf16 elems
                tdm_load_tile(cur ? hiOff0 : hiOff1, cbhi + goff);
                tdm_load_tile(cur ? loOff0 : loOff1, cblo + goff);
                __builtin_amdgcn_s_wait_tensorcnt(2);   // tile t landed (in-order), t+1 in flight
            } else {
                __builtin_amdgcn_s_wait_tensorcnt(0);
            }
        }
        if (tid < KT) sCn[tid] = cnorm[t * KT + tid];
        __syncthreads();                                // publish DMA'd tile to all waves

        compute_tile(sBhi[cur], sBlo[cur], sCn, Ahi, Alo, lane, t * KT, st);
        __syncthreads();                                // tile consumed; buffer reusable
    }

    finish_block(z, cb, zq_out, idx_out, partials, sIdx, sRed, tid, lane, w, row0, st);
}

// ---------------- main kernel, fallback path: convert-in-kernel staging -----------------
__global__ void __launch_bounds__(256)
vq_main_kernel(const float* __restrict__ z,
               const float* __restrict__ cb,
               const float* __restrict__ cnorm,
               float* __restrict__ zq_out,
               int*   __restrict__ idx_out,
               float* __restrict__ partials) {
    __shared__ unsigned short sBhi[KT * LROW];
    __shared__ unsigned short sBlo[KT * LROW];
    __shared__ float sCn[KT];
    __shared__ int   sIdx[ROWS_BLK];
    __shared__ float sRed[256];

    const int tid  = threadIdx.x;
    const int lane = tid & 31;
    const int w    = tid >> 5;
    const int row0 = blockIdx.x * ROWS_BLK;

    BF16x16 Ahi[DCHUNKS], Alo[DCHUNKS];
    build_A_frags(z, row0, w, lane, Ahi, Alo);

    ArgminState st;
    #pragma unroll
    for (int r = 0; r < 8; ++r) { st.best[r] = 3.402823466e+38f; st.bidx[r] = 0; }

    for (int k0 = 0; k0 < KCODES; k0 += KT) {
        #pragma unroll
        for (int it = 0; it < (KT * DIM / 4) / 256; ++it) {
            int flat4 = it * 256 + tid;
            int row   = flat4 >> 6;
            int dim4  = flat4 & 63;
            float4 v  = ((const float4*)(cb + (size_t)(k0 + row) * DIM))[dim4];
            unsigned short h0, h1, h2, h3, l0, l1, l2, l3;
            splitf(v.x, h0, l0); splitf(v.y, h1, l1);
            splitf(v.z, h2, l2); splitf(v.w, h3, l3);
            int o = row * LROW + dim4 * 4;
            *(uint2*)&sBhi[o] = make_uint2((unsigned)h0 | ((unsigned)h1 << 16),
                                           (unsigned)h2 | ((unsigned)h3 << 16));
            *(uint2*)&sBlo[o] = make_uint2((unsigned)l0 | ((unsigned)l1 << 16),
                                           (unsigned)l2 | ((unsigned)l3 << 16));
        }
        if (tid < KT) sCn[tid] = cnorm[k0 + tid];
        __syncthreads();

        compute_tile(sBhi, sBlo, sCn, Ahi, Alo, lane, k0, st);
        __syncthreads();
    }

    finish_block(z, cb, zq_out, idx_out, partials, sIdx, sRed, tid, lane, w, row0, st);
}

// ---------------- loss finalize (deterministic fixed-order sum) ----------------
__global__ void vq_finalize_kernel(const float* __restrict__ partials, float* __restrict__ lossp) {
    __shared__ float sRed[256];
    int t = threadIdx.x;
    sRed[t] = partials[t];
    __syncthreads();
    #pragma unroll
    for (int s = 128; s > 0; s >>= 1) {
        if (t < s) sRed[t] += sRed[t + s];
        __syncthreads();
    }
    if (t == 0) {
        float mean = sRed[0] / (float)(NTOK * DIM);
        *lossp = mean + 0.25f * mean;       // diff2 + COMMITMENT_COST * diff2
    }
}

extern "C" void kernel_launch(void* const* d_in, const int* in_sizes, int n_in,
                              void* d_out, int out_size, void* d_ws, size_t ws_size,
                              hipStream_t stream) {
    (void)in_sizes; (void)n_in; (void)out_size;
    const float* z  = (const float*)d_in[0];     // [16,2048,256] f32
    const float* cb = (const float*)d_in[1];     // [8192,256]    f32

    float* zq      = (float*)d_out;
    int*   idx_out = (int*)d_out + (size_t)NTOK * DIM;
    float* lossp   = (float*)d_out + (size_t)NTOK * DIM + NTOK;

    float* cnorm    = (float*)d_ws;                       // 8192 floats
    float* partials = cnorm + KCODES;                     // 256 floats
    unsigned short* cbhi = (unsigned short*)((char*)d_ws + (KCODES + 256) * sizeof(float));
    unsigned short* cblo = cbhi + (size_t)KCODES * DIM;

    const size_t ws_need = (size_t)(KCODES + 256) * sizeof(float)
                         + (size_t)KCODES * DIM * 2 * sizeof(unsigned short);

    vq_cnorm_kernel<<<KCODES / 256, 256, 0, stream>>>(cb, cnorm);
    if (ws_size >= ws_need) {
        vq_prep_kernel<<<(KCODES * DIM / 4) / 256, 256, 0, stream>>>(cb, cbhi, cblo);
        vq_main_tdm_kernel<<<NTOK / ROWS_BLK, 256, 0, stream>>>(z, cb, cbhi, cblo, cnorm,
                                                                zq, idx_out, partials);
    } else {
        vq_main_kernel<<<NTOK / ROWS_BLK, 256, 0, stream>>>(z, cb, cnorm,
                                                            zq, idx_out, partials);
    }
    vq_finalize_kernel<<<1, 256, 0, stream>>>(partials, lossp);
}
